// CharMIRNN_23545010717002
// MI455X (gfx1250) — matrix operand order repaired and checked by reading, not compiled, not run on hardware
//
#include <hip/hip_runtime.h>
#include <hip/hip_bf16.h>
#include <math.h>

// Problem dims (fixed by the reference)
#define B_ 64
#define L_ 512
#define V_ 256
#define E_ 512
#define H_ 1024

typedef __bf16 bf16_t;
typedef __attribute__((ext_vector_type(16))) __bf16 v16bf;
typedef __attribute__((ext_vector_type(8)))  __bf16 v8bf;
typedef __attribute__((ext_vector_type(8)))  float  v8f;

// f32 -> bf16 round-to-nearest-even via integer trick (no intrinsic dependence)
static __device__ __forceinline__ bf16_t to_bf16(float f) {
  union { float f; unsigned u; } in; in.f = f;
  unsigned r = in.u + 0x7FFFu + ((in.u >> 16) & 1u);
  union { unsigned short s; bf16_t b; } out; out.s = (unsigned short)(r >> 16);
  return out.b;
}

// ---- WMMA fragment loaders (CDNA5 wave32 layouts, CDNA5 ISA §7.12.2) ----
// A: 16x32 bf16 tile, row-major source, row m = lane%16.
//   lanes 0-15 : elems 0..7 = K 0..7,   elems 8..15 = K 16..23
//   lanes 16-31: elems 0..7 = K 8..15,  elems 8..15 = K 24..31
static __device__ __forceinline__ v16bf load_a_frag(const bf16_t* __restrict__ A,
                                                    int lda, int kbase, int lane) {
  int half = lane >> 4;
  const bf16_t* p = A + (size_t)(lane & 15) * lda + kbase + 8 * half;
  v8bf lo = *(const v8bf*)(p);
  v8bf hi = *(const v8bf*)(p + 16);
  v16bf r;
#pragma unroll
  for (int i = 0; i < 8; ++i) { r[i] = lo[i]; r[i + 8] = hi[i]; }
  return r;
}

// B: 32x16 bf16 tile taken from W = B^T stored row-major [N][K] (weight matrix [out][in]).
//   lane n = lane%16; same K order as A:
//   lanes 0-15 : elems 0..7 = K 0..7,   elems 8..15 = K 16..23
//   lanes 16-31: elems 0..7 = K 8..15,  elems 8..15 = K 24..31
static __device__ __forceinline__ v16bf load_b_frag(const bf16_t* __restrict__ W,
                                                    int ldw, int kbase, int lane) {
  int half = lane >> 4;
  const bf16_t* p = W + (size_t)(lane & 15) * ldw + kbase + 8 * half;
  v8bf lo = *(const v8bf*)(p);
  v8bf hi = *(const v8bf*)(p + 16);
  v16bf r;
#pragma unroll
  for (int i = 0; i < 8; ++i) { r[i] = lo[i]; r[i + 8] = hi[i]; }
  return r;
}

// ---- kernels ----

__global__ void cvt_bf16_kernel(const float* __restrict__ src, bf16_t* __restrict__ dst, int n) {
  int i = blockIdx.x * blockDim.x + threadIdx.x;
  if (i < n) dst[i] = to_bf16(src[i]);
}

// Uemb[V, H] = emb[V, E] @ U_w[H, E]^T   (tiny GEMM; folds embedding+projection)
__global__ void uemb_gemm_kernel(const bf16_t* __restrict__ embb,
                                 const bf16_t* __restrict__ Uwb,
                                 float* __restrict__ Uemb) {
  int lane = threadIdx.x & 31;
  int g = blockIdx.x * (blockDim.x >> 5) + (threadIdx.x >> 5);
  int mtile = g & 15;   // 256/16 = 16 row tiles
  int ntile = g >> 4;   // 1024/16 = 64 col tiles
  const bf16_t* A = embb + (size_t)mtile * 16 * E_;
  const bf16_t* W = Uwb  + (size_t)ntile * 16 * E_;
  v8f acc = {};
  for (int k = 0; k < E_; k += 32) {
    v16bf a = load_a_frag(A, E_, k, lane);
    v16bf b = load_b_frag(W, E_, k, lane);
    acc = __builtin_amdgcn_wmma_f32_16x16x32_bf16(false, a, false, b, (short)0, acc, false, false);
  }
  int half = lane >> 4;
  int n = ntile * 16 + (lane & 15);
#pragma unroll
  for (int r = 0; r < 8; ++r) {
    int m = mtile * 16 + half * 8 + r;
    Uemb[(size_t)m * H_ + n] = acc[r];
  }
}

// One recurrence step: Vh = h@V_w^T ; h' = tanh(alpha*Vh*ux + b1*Vh + b2*ux + bias)
// ux gathered from the precomputed Uemb table. Double-buffered h (hin/hout).
__global__ void rnn_step_kernel(int t,
                                const bf16_t* __restrict__ hin,
                                bf16_t* __restrict__ hout,
                                const bf16_t* __restrict__ Vwb,
                                const float* __restrict__ Uemb,
                                const int* __restrict__ x,
                                const float* __restrict__ alpha,
                                const float* __restrict__ beta1,
                                const float* __restrict__ beta2,
                                const float* __restrict__ bias,
                                bf16_t* __restrict__ hs,
                                float* __restrict__ hlast) {
  int lane = threadIdx.x & 31;
  int g = blockIdx.x * (blockDim.x >> 5) + (threadIdx.x >> 5);
  int mtile = g & 3;    // 64/16 = 4 row tiles (batch)
  int ntile = g >> 2;   // 1024/16 = 64 col tiles (hidden)
  const bf16_t* A = hin + (size_t)mtile * 16 * H_;
  const bf16_t* W = Vwb + (size_t)ntile * 16 * H_;
  v8f acc = {};
#pragma unroll 4
  for (int k = 0; k < H_; k += 32) {
    v16bf a = load_a_frag(A, H_, k, lane);
    v16bf b = load_b_frag(W, H_, k, lane);
    acc = __builtin_amdgcn_wmma_f32_16x16x32_bf16(false, a, false, b, (short)0, acc, false, false);
  }
  int half = lane >> 4;
  int n = ntile * 16 + (lane & 15);
  float al = alpha[n], b1 = beta1[n], b2 = beta2[n], bi = bias[n];
#pragma unroll
  for (int r = 0; r < 8; ++r) {
    int b = mtile * 16 + half * 8 + r;       // batch row
    int tok = x[(size_t)b * L_ + t];
    float ux = Uemb[(size_t)tok * H_ + n];
    float vh = acc[r];
    float m  = al * (vh * ux) + b1 * vh + b2 * ux + bi;
    float h  = tanhf(m);
    bf16_t hb = to_bf16(h);
    hout[(size_t)b * H_ + n] = hb;                        // state for next step
    hs[((size_t)b * L_ + t) * H_ + n] = hb;               // decoder operand
    if (t == L_ - 1) hlast[(size_t)b * H_ + n] = h;       // h_last output (f32)
  }
}

// logits[B*L, V] = hs[B*L, H] @ dec_w[V, H]^T + dec_b ; 4 N-tiles per wave for A reuse
__global__ void decoder_kernel(const bf16_t* __restrict__ hs,
                               const bf16_t* __restrict__ decwb,
                               const float* __restrict__ dec_b,
                               float* __restrict__ out) {
  int lane = threadIdx.x & 31;
  int g = blockIdx.x * (blockDim.x >> 5) + (threadIdx.x >> 5);
  int ngrp  = g & 3;    // 4 groups x 4 tiles = 16 col tiles (V=256)
  int mtile = g >> 2;   // 32768/16 = 2048 row tiles
  const bf16_t* A = hs + (size_t)mtile * 16 * H_;
  v8f acc[4] = {};
  for (int k = 0; k < H_; k += 32) {
    v16bf a = load_a_frag(A, H_, k, lane);
#pragma unroll
    for (int j = 0; j < 4; ++j) {
      const bf16_t* W = decwb + (size_t)(ngrp * 4 + j) * 16 * H_;
      v16bf b = load_b_frag(W, H_, k, lane);
      acc[j] = __builtin_amdgcn_wmma_f32_16x16x32_bf16(false, a, false, b, (short)0, acc[j], false, false);
    }
  }
  int half = lane >> 4;
#pragma unroll
  for (int j = 0; j < 4; ++j) {
    int n = (ngrp * 4 + j) * 16 + (lane & 15);
    float db = dec_b[n];
#pragma unroll
    for (int r = 0; r < 8; ++r) {
      int i = mtile * 16 + half * 8 + r;
      out[(size_t)i * V_ + n] = acc[j][r] + db;
    }
  }
}

extern "C" void kernel_launch(void* const* d_in, const int* in_sizes, int n_in,
                              void* d_out, int out_size, void* d_ws, size_t ws_size,
                              hipStream_t stream) {
  (void)in_sizes; (void)n_in; (void)out_size; (void)ws_size;

  const int*   x     = (const int*)  d_in[0];
  const float* emb   = (const float*)d_in[1];
  const float* U_w   = (const float*)d_in[2];
  const float* V_w   = (const float*)d_in[3];
  const float* alpha = (const float*)d_in[4];
  const float* beta1 = (const float*)d_in[5];
  const float* beta2 = (const float*)d_in[6];
  const float* bias  = (const float*)d_in[7];
  const float* dec_w = (const float*)d_in[8];
  const float* dec_b = (const float*)d_in[9];

  float* logits = (float*)d_out;
  float* hlast  = logits + (size_t)B_ * L_ * V_;   // [1,B,H] tail of d_out

  // Workspace carve (~69 MB total)
  char* w = (char*)d_ws;
  float*  Uemb  = (float*)w;  w += (size_t)V_ * H_ * sizeof(float);    // 1 MB
  bf16_t* Vwb   = (bf16_t*)w; w += (size_t)H_ * H_ * sizeof(bf16_t);   // 2 MB
  bf16_t* decwb = (bf16_t*)w; w += (size_t)V_ * H_ * sizeof(bf16_t);   // 0.5 MB
  bf16_t* embb  = (bf16_t*)w; w += (size_t)V_ * E_ * sizeof(bf16_t);   // 0.25 MB
  bf16_t* Uwb   = (bf16_t*)w; w += (size_t)H_ * E_ * sizeof(bf16_t);   // 1 MB
  bf16_t* hbuf0 = (bf16_t*)w; w += (size_t)B_ * H_ * sizeof(bf16_t);   // 128 KB
  bf16_t* hbuf1 = (bf16_t*)w; w += (size_t)B_ * H_ * sizeof(bf16_t);   // 128 KB
  bf16_t* hs    = (bf16_t*)w; w += (size_t)B_ * L_ * H_ * sizeof(bf16_t); // 64 MB
  bf16_t* hbuf[2] = { hbuf0, hbuf1 };

  // 1) weight conversions to bf16 + zero-init h0
  cvt_bf16_kernel<<<(V_*E_ + 255) / 256, 256, 0, stream>>>(emb,   embb,  V_*E_);
  cvt_bf16_kernel<<<(H_*E_ + 255) / 256, 256, 0, stream>>>(U_w,   Uwb,   H_*E_);
  cvt_bf16_kernel<<<(H_*H_ + 255) / 256, 256, 0, stream>>>(V_w,   Vwb,   H_*H_);
  cvt_bf16_kernel<<<(V_*H_ + 255) / 256, 256, 0, stream>>>(dec_w, decwb, V_*H_);
  hipMemsetAsync(hbuf[0], 0, (size_t)B_ * H_ * sizeof(bf16_t), stream);

  // 2) Uemb = emb @ U_w^T  (folds the entire 34 GFLOP input projection into a 1 MB table)
  uemb_gemm_kernel<<<128, 256, 0, stream>>>(embb, Uwb, Uemb);

  // 3) sequential recurrence: 512 small fused WMMA kernels (graph-replayed)
  for (int t = 0; t < L_; ++t) {
    rnn_step_kernel<<<32, 256, 0, stream>>>(t, hbuf[t & 1], hbuf[(t + 1) & 1],
                                            Vwb, Uemb, x, alpha, beta1, beta2, bias,
                                            hs, hlast);
  }

  // 4) decoder GEMM + bias
  decoder_kernel<<<1024, 256, 0, stream>>>(hs, decwb, dec_b, logits);
}
